// TemporalEncoding_copy_78950088835529
// MI455X (gfx1250) — compile-verified, matrix operand order checked by reading, not run on hardware
//
#include <hip/hip_runtime.h>

// D = in_features = 512 (fixed by the reference model)
#define DD 512

typedef __attribute__((ext_vector_type(16))) __bf16 v16bf;
typedef __attribute__((ext_vector_type(8)))  __bf16 v8bf;
typedef __attribute__((ext_vector_type(8)))  float  v8f;

// fp32 -> bf16, round-to-nearest-even (bit-level; no reliance on native cvt)
__device__ __forceinline__ __bf16 f2bf(float f) {
  union { float f; unsigned u; } x; x.f = f;
  unsigned u = x.u;
  u += 0x7FFFu + ((u >> 16) & 1u);
  union { unsigned short s; __bf16 b; } y; y.s = (unsigned short)(u >> 16);
  return y.b;
}

__device__ __forceinline__ void cvt8(const float4& a, const float4& b, v8bf& p) {
  p[0]=f2bf(a.x); p[1]=f2bf(a.y); p[2]=f2bf(a.z); p[3]=f2bf(a.w);
  p[4]=f2bf(b.x); p[5]=f2bf(b.y); p[6]=f2bf(b.z); p[7]=f2bf(b.w);
}

// ---------------------------------------------------------------------------
// Weight prep: fp32 [K,N] row-major -> bf16 WT[n*K + k]  (B-operand friendly:
// one lane's K-contiguous 8 bf16 = one 16B global_load_b128)
// ---------------------------------------------------------------------------
__global__ __launch_bounds__(256)
void k_cvt_w_transpose(const float* __restrict__ W, __bf16* __restrict__ WT) {
  const int id = blockIdx.x * 256 + threadIdx.x;   // 0 .. 512*512-1
  const int n = id >> 9;
  const int k = id & (DD - 1);
  WT[(size_t)n * DD + k] = f2bf(W[(size_t)k * DD + n]);
}

// lin_w is [D, 2D] row-major == already B^T layout for g @ lin_w.T; just cast.
__global__ __launch_bounds__(256)
void k_cvt_lin(const float* __restrict__ Wl, __bf16* __restrict__ LT) {
  const int id = blockIdx.x * 256 + threadIdx.x;   // 0 .. 512*1024-1
  LT[id] = f2bf(Wl[id]);
}

// ---------------------------------------------------------------------------
// GEMM1: C[rows,512] = A[rows,512](f32) @ W(bf16, pre-transposed WT[n*512+k])
// gridDim.z selects (h_p, W_o) vs (h_k, W_c).
// Block: 256 thr (8 waves), tile 128(M) x 128(N). Wave = 32x64 -> 2x4 WMMA acc.
// Double-buffered LDS A staging: one barrier per K-step; next slice's global
// loads are issued before the current slice's WMMAs (latency hidden by XDL).
// ---------------------------------------------------------------------------
__global__ __launch_bounds__(256)
void k_wmma_gemm512(const float* __restrict__ A0, const float* __restrict__ A1,
                    const __bf16* __restrict__ WT0, const __bf16* __restrict__ WT1,
                    float* __restrict__ Cbase, int rowsPerMat)
{
  __shared__ __align__(16) __bf16 sA[2][128 * 40];  // 2 x (128 rows, 80B stride)

  const int z = blockIdx.z;
  const float*  A  = z ? A1  : A0;
  const __bf16* WT = z ? WT1 : WT0;
  float* C = Cbase + (size_t)z * rowsPerMat * DD;

  const int tid  = threadIdx.x;
  const int lane = tid & 31;
  const int w    = tid >> 5;
  const int wm   = w & 3;          // 4 wave rows -> M offset 32*wm
  const int wn   = w >> 2;         // 2 wave cols -> N offset 64*wn
  const int lr   = lane & 15;      // row/col within 16x16 tile
  const int lh   = lane >> 4;      // lane half selects K sub-block

  const int bm = blockIdx.y * 128;
  const int bn = blockIdx.x * 128;

  v8f acc[2][4];
#pragma unroll
  for (int i = 0; i < 2; ++i)
#pragma unroll
    for (int j = 0; j < 4; ++j) acc[i][j] = 0.0f;

  const int srow  = tid >> 1;      // staging: 2 threads per row, 16 floats each
  const int shalf = tid & 1;
  const float* srcBase = A + (size_t)(bm + srow) * DD + shalf * 16;
  __bf16* const sDst = &sA[0][0] + srow * 40 + shalf * 16;  // buffer-0 slot

  // ---- prologue: stage K-slice 0 into buffer 0 ----
  {
    const float4* s4 = (const float4*)srcBase;
    v8bf p0, p1;
    cvt8(s4[0], s4[1], p0);
    cvt8(s4[2], s4[3], p1);
    *(v8bf*)sDst       = p0;
    *(v8bf*)(sDst + 8) = p1;
  }

  int cur = 0;
  for (int kt = 0; kt < DD; kt += 32) {
    __syncthreads();                       // buffer `cur` ready for all waves

    // ---- issue next K-slice's global loads early ----
    const bool more = (kt + 32) < DD;
    float4 g0 = {}, g1 = {}, g2 = {}, g3 = {};
    if (more) {
      const float4* s4 = (const float4*)(srcBase + kt + 32);
      g0 = s4[0]; g1 = s4[1]; g2 = s4[2]; g3 = s4[3];
      __builtin_prefetch(srcBase + kt + 64, 0, 3);
    }

    // ---- A fragments (ISA 16-bit A layout: lanes<16 K=0..7/16..23, else 8..15/24..31)
    const __bf16* sBuf = &sA[cur][0];
    v16bf a[2];
#pragma unroll
    for (int i = 0; i < 2; ++i) {
      const __bf16* p = sBuf + (wm * 32 + i * 16 + lr) * 40 + lh * 8;
      v8bf lo = *(const v8bf*)p;          // K = lh*8 .. +7
      v8bf hi = *(const v8bf*)(p + 16);   // K = 16 + lh*8 .. +7
#pragma unroll
      for (int e = 0; e < 8; ++e) { a[i][e] = lo[e]; a[i][e + 8] = hi[e]; }
    }

    // ---- B fragments (b128 from L2-resident bf16 WT) + WMMA ----
#pragma unroll
    for (int j = 0; j < 4; ++j) {
      const __bf16* q = WT + (size_t)(bn + wn * 64 + j * 16 + lr) * DD + kt + lh * 8;
      v8bf lo = *(const v8bf*)q;
      v8bf hi = *(const v8bf*)(q + 16);
      v16bf b;
#pragma unroll
      for (int e = 0; e < 8; ++e) { b[e] = lo[e]; b[e + 8] = hi[e]; }
#pragma unroll
      for (int i = 0; i < 2; ++i)
        acc[i][j] = __builtin_amdgcn_wmma_f32_16x16x32_bf16(
            false, a[i], false, b, (short)0, acc[i][j], false, false);
    }

    // ---- store next slice into the opposite buffer (no barrier needed:
    //      top-of-loop barrier ensures its last readers are done) ----
    if (more) {
      v8bf p0, p1;
      cvt8(g0, g1, p0);
      cvt8(g2, g3, p1);
      __bf16* d = sDst + (cur ^ 1) * (128 * 40);
      *(v8bf*)d       = p0;
      *(v8bf*)(d + 8) = p1;
    }
    cur ^= 1;
  }

  // ---- epilogue (C/D layout: M = vgpr + 8*(lane>=16), N = lane%16) ----
#pragma unroll
  for (int i = 0; i < 2; ++i) {
    const int mbase = bm + wm * 32 + i * 16 + lh * 8;
#pragma unroll
    for (int j = 0; j < 4; ++j) {
      const int n = bn + wn * 64 + j * 16 + lr;
#pragma unroll
      for (int v = 0; v < 8; ++v)
        C[(size_t)(mbase + v) * DD + n] = acc[i][j][v];
    }
  }
}

// ---------------------------------------------------------------------------
// GEMM2 (fused mutual stage):
//   out = leaky_relu(tanh([thp[ip], thk[ik]]) @ lin_w.T + lin_b) + bias
//   scatter into thp/thk rows (in place).
// Block owns full N=512 (BN=512) so its read rows == its write rows -> no
// cross-block RAW hazard for unique indices. Block: 64(M) x 512(N), 8 waves
// as 2(M) x 4(N); wave = 32x128 -> 2x8 WMMA accumulators. K = 1024.
// Double-buffered LDS, tanh fused into staging.
// ---------------------------------------------------------------------------
__global__ __launch_bounds__(256)
void k_wmma_mutual(const float* __restrict__ thp,
                   const float* __restrict__ thk,
                   const int*   __restrict__ idxP,
                   const int*   __restrict__ idxK,
                   const __bf16* __restrict__ LT,   // bf16 lin_w [512][1024]
                   const float* __restrict__ linB,
                   const float* __restrict__ bias,
                   float* __restrict__ outP,
                   float* __restrict__ outK)
{
  __shared__ __align__(16) __bf16 sA[2][64 * 40];
  __shared__ int sIp[64];
  __shared__ int sIk[64];

  const int tid  = threadIdx.x;
  const int lane = tid & 31;
  const int w    = tid >> 5;
  const int wm   = w & 1;          // M offset 32*wm
  const int wn   = w >> 1;         // N offset 128*wn
  const int lr   = lane & 15;
  const int lh   = lane >> 4;

  const int bm = blockIdx.x * 64;
  if (tid < 64) { sIp[tid] = idxP[bm + tid]; sIk[tid] = idxK[bm + tid]; }
  __syncthreads();

  v8f acc[2][8];
#pragma unroll
  for (int i = 0; i < 2; ++i)
#pragma unroll
    for (int j = 0; j < 8; ++j) acc[i][j] = 0.0f;

  const int srow = tid >> 2;       // staging: 4 threads per row, 8 floats each
  const int sq   = tid & 3;
  const float* rowP = thp + (size_t)sIp[srow] * DD;
  const float* rowK = thk + (size_t)sIk[srow] * DD;
  __bf16* const sDst = &sA[0][0] + srow * 40 + sq * 8;

  // gathered row segment for K-slice kt (8-chunks never straddle the 512 split)
  auto srcAt = [&](int kt) -> const float4* {
    const int k = kt + sq * 8;
    return (const float4*)((k < DD) ? (rowP + k) : (rowK + (k - DD)));
  };

  // ---- prologue: stage K-slice 0 ----
  {
    const float4* s4 = srcAt(0);
    float4 f0 = s4[0], f1 = s4[1];
    v8bf p;
    p[0]=f2bf(tanhf(f0.x)); p[1]=f2bf(tanhf(f0.y));
    p[2]=f2bf(tanhf(f0.z)); p[3]=f2bf(tanhf(f0.w));
    p[4]=f2bf(tanhf(f1.x)); p[5]=f2bf(tanhf(f1.y));
    p[6]=f2bf(tanhf(f1.z)); p[7]=f2bf(tanhf(f1.w));
    *(v8bf*)sDst = p;
  }

  int cur = 0;
  for (int kt = 0; kt < 2 * DD; kt += 32) {
    __syncthreads();

    const bool more = (kt + 32) < 2 * DD;
    float4 g0 = {}, g1 = {};
    if (more) {
      const float4* s4 = srcAt(kt + 32);
      g0 = s4[0]; g1 = s4[1];
    }

    const __bf16* sBuf = &sA[cur][0];
    v16bf a[2];
#pragma unroll
    for (int i = 0; i < 2; ++i) {
      const __bf16* p = sBuf + (wm * 32 + i * 16 + lr) * 40 + lh * 8;
      v8bf lo = *(const v8bf*)p;
      v8bf hi = *(const v8bf*)(p + 16);
#pragma unroll
      for (int e = 0; e < 8; ++e) { a[i][e] = lo[e]; a[i][e + 8] = hi[e]; }
    }

#pragma unroll
    for (int j = 0; j < 8; ++j) {
      const __bf16* q = LT + (size_t)(wn * 128 + j * 16 + lr) * (2 * DD) + kt + lh * 8;
      v8bf lo = *(const v8bf*)q;
      v8bf hi = *(const v8bf*)(q + 16);
      v16bf b;
#pragma unroll
      for (int e = 0; e < 8; ++e) { b[e] = lo[e]; b[e + 8] = hi[e]; }
#pragma unroll
      for (int i = 0; i < 2; ++i)
        acc[i][j] = __builtin_amdgcn_wmma_f32_16x16x32_bf16(
            false, a[i], false, b, (short)0, acc[i][j], false, false);
    }

    if (more) {
      v8bf p;
      p[0]=f2bf(tanhf(g0.x)); p[1]=f2bf(tanhf(g0.y));
      p[2]=f2bf(tanhf(g0.z)); p[3]=f2bf(tanhf(g0.w));
      p[4]=f2bf(tanhf(g1.x)); p[5]=f2bf(tanhf(g1.y));
      p[6]=f2bf(tanhf(g1.z)); p[7]=f2bf(tanhf(g1.w));
      *(v8bf*)(sDst + (cur ^ 1) * (64 * 40)) = p;
    }
    cur ^= 1;
  }

  // ---- epilogue: +lin_b, leaky_relu(0.01), +bias, scatter to both outputs ----
#pragma unroll
  for (int i = 0; i < 2; ++i) {
    const int mloc = wm * 32 + i * 16 + lh * 8;
#pragma unroll
    for (int j = 0; j < 8; ++j) {
      const int n = wn * 128 + j * 16 + lr;
      const float lb = linB[n];
      const float bs = bias[n];
#pragma unroll
      for (int v = 0; v < 8; ++v) {
        float r = acc[i][j][v] + lb;
        r = (r > 0.0f) ? r : 0.01f * r;
        r += bs;
        const int ml = mloc + v;
        outP[(size_t)sIp[ml] * DD + n] = r;
        outK[(size_t)sIk[ml] * DD + n] = r;
      }
    }
  }
}

// ---------------------------------------------------------------------------
extern "C" void kernel_launch(void* const* d_in, const int* in_sizes, int n_in,
                              void* d_out, int out_size, void* d_ws, size_t ws_size,
                              hipStream_t stream) {
  (void)n_in; (void)out_size; (void)ws_size;

  const float* h_p   = (const float*)d_in[0];
  const float* h_k   = (const float*)d_in[1];
  const int*   idxP  = (const int*)d_in[2];   // jax default: int32
  const int*   idxK  = (const int*)d_in[3];
  /* d_in[4] = last_x: unused by the reference */
  const float* w_o   = (const float*)d_in[5];
  const float* w_c   = (const float*)d_in[6];
  const float* lin_w = (const float*)d_in[7];
  const float* lin_b = (const float*)d_in[8];
  const float* bias  = (const float*)d_in[9];

  const int N = in_sizes[0] / DD;   // 65536
  const int M = in_sizes[2];        // 32768

  float* out_hp = (float*)d_out;
  float* out_hk = out_hp + (size_t)N * DD;

  // workspace: bf16 weights (2 MB total)
  __bf16* wtO  = (__bf16*)d_ws;
  __bf16* wtC  = wtO + (size_t)DD * DD;
  __bf16* linT = wtC + (size_t)DD * DD;

  k_cvt_w_transpose<<<(DD * DD) / 256, 256, 0, stream>>>(w_o, wtO);
  k_cvt_w_transpose<<<(DD * DD) / 256, 256, 0, stream>>>(w_c, wtC);
  k_cvt_lin<<<(DD * 2 * DD) / 256, 256, 0, stream>>>(lin_w, linT);

  dim3 g1(DD / 128, N / 128, 2);    // (4, 512, 2): z selects (h_p,W_o)/(h_k,W_c)
  k_wmma_gemm512<<<g1, 256, 0, stream>>>(h_p, h_k, wtO, wtC, out_hp, N);

  k_wmma_mutual<<<M / 64, 256, 0, stream>>>(out_hp, out_hk, idxP, idxK,
                                            linT, lin_b, bias, out_hp, out_hk);
}
